// BBRNNModel_43404939493527
// MI455X (gfx1250) — matrix operand order repaired and checked by reading, not compiled, not run on hardware
//
#include <hip/hip_runtime.h>

// Bidirectional GRU (reset_after, relu/sigmoid) + per-step Dense + softmax.
// B=32, T=2048, F=U=128, C=11.  gfx1250 wave32, bf16 WMMA with f32 accum.

#define B_  32
#define T_  2048
#define F_  128
#define U_  128
#define C_  11
#define G3  384          // 3*U  (gate order z, r, h)
#define MROWS (B_*T_)    // 65536

typedef __attribute__((ext_vector_type(16))) __bf16 v16bf;
typedef __attribute__((ext_vector_type(8)))  float  v8f;

__device__ __forceinline__ v8f zero8() {
    v8f z;
#pragma unroll
    for (int i = 0; i < 8; ++i) z[i] = 0.0f;
    return z;
}

__device__ __forceinline__ float sigmoidf_(float x) {
    return 1.0f / (1.0f + __expf(-x));
}

// ---- WMMA fragment builders (ISA 7.12.2 layouts, wave32) -------------------
// A: 16x32 bf16, source row-major [m][k] (bf16), row stride ld (elements).
// lanes 0-15: m=lane, K = {0..7} in halves 0..7, {16..23} in halves 8..15
// lanes 16-31: m=lane-16, K = {8..15} and {24..31}
__device__ __forceinline__ v16bf frag_a_bf16(const __bf16* __restrict__ s, int ld) {
    const int lane  = threadIdx.x & 31;
    const int m     = lane & 15;
    const int kbase = (lane < 16) ? 0 : 8;
    v16bf f;
#pragma unroll
    for (int v = 0; v < 4; ++v) {
        f[2*v]     = s[m*ld + kbase + 2*v];
        f[2*v + 1] = s[m*ld + kbase + 2*v + 1];
        f[8 + 2*v]     = s[m*ld + kbase + 16 + 2*v];
        f[8 + 2*v + 1] = s[m*ld + kbase + 16 + 2*v + 1];
    }
    return f;
}

// B: 32x16 bf16 built from a TRANSPOSED weight image [n][k] (k contiguous),
// row stride ld. lanes 0-15: n=lane, K={0..7,16..23}; lanes 16-31: n=lane-16, K={8..15,24..31}.
__device__ __forceinline__ v16bf frag_b_tr(const __bf16* __restrict__ s, int ld) {
    const int lane  = threadIdx.x & 31;
    const int n     = lane & 15;
    const int kbase = (lane < 16) ? 0 : 8;
    v16bf f;
#pragma unroll
    for (int v = 0; v < 4; ++v) {
        f[2*v]     = s[n*ld + kbase + 2*v];
        f[2*v + 1] = s[n*ld + kbase + 2*v + 1];
        f[8 + 2*v]     = s[n*ld + kbase + 16 + 2*v];
        f[8 + 2*v + 1] = s[n*ld + kbase + 16 + 2*v + 1];
    }
    return f;
}

#define WMMA_BF16(A, Bf, Cacc) \
    __builtin_amdgcn_wmma_f32_16x16x32_bf16(false, (A), false, (Bf), (short)0, (Cacc), false, false)

// ---------------------------------------------------------------------------
// Kernel 1: convert fp32 weight [K][N] -> bf16 transposed [N][K]
__global__ void cvt_tr_kernel(const float* __restrict__ W, __bf16* __restrict__ Wt,
                              int K, int N) {
    int idx = blockIdx.x * blockDim.x + threadIdx.x;
    if (idx >= K * N) return;
    int k = idx / N, n = idx - k * N;          // coalesced read over n
    Wt[n * K + k] = (__bf16)W[idx];
}

// ---------------------------------------------------------------------------
// Kernel 2: xg[t][b][g] = x @ W + b0 (time-major output for the scan).
// x:[65536][128] f32 (row = b*T+t), Wt:[384][128] bf16(T), xg:[T][B][384] f32.
// Block = 256 thr (8 waves) computes 64(M) x 128(N).
__global__ __launch_bounds__(256) void proj_kernel(
    const float* __restrict__ x, const __bf16* __restrict__ Wt,
    const float* __restrict__ b0, float* __restrict__ xg) {
    __shared__ __bf16 lds_a[64][F_ + 8];       // x tile, bf16, padded

    const int m0 = blockIdx.x * 64;
    const int n0 = blockIdx.y * 128;

    // stage + convert A tile (coalesced)
    for (int i = threadIdx.x; i < 64 * F_; i += 256) {
        int r = i >> 7, c = i & 127;
        lds_a[r][c] = (__bf16)x[(size_t)(m0 + r) * F_ + c];
    }
    __syncthreads();

    const int wave = threadIdx.x >> 5;
    const int rt   = wave & 3;                 // row tile 0..3  (16 rows each)
    const int cg   = (wave >> 2) * 4;          // col tile base 0 or 4

    v8f acc[4];
#pragma unroll
    for (int j = 0; j < 4; ++j) acc[j] = zero8();

#pragma unroll
    for (int kc = 0; kc < 4; ++kc) {           // K = 128 = 4 x 32
        v16bf a = frag_a_bf16(&lds_a[rt * 16][kc * 32], F_ + 8);
#pragma unroll
        for (int j = 0; j < 4; ++j) {
            int n = n0 + (cg + j) * 16;
            v16bf bf = frag_b_tr(Wt + (size_t)n * F_ + kc * 32, F_);
            acc[j] = WMMA_BF16(a, bf, acc[j]);
        }
    }

    const int lane  = threadIdx.x & 31;
    const int ncol  = lane & 15;
    const int mhalf = (lane >> 4) * 8;
#pragma unroll
    for (int j = 0; j < 4; ++j) {
        int n = n0 + (cg + j) * 16 + ncol;
        float bias = b0[n];
#pragma unroll
        for (int r = 0; r < 8; ++r) {
            int m = m0 + rt * 16 + mhalf + r;  // m = b*T + t
            int tt = m & (T_ - 1);
            int bb = m >> 11;                  // T = 2048 = 2^11
            xg[((size_t)tt * B_ + bb) * G3 + n] = acc[j][r] + bias;
        }
    }
}

// ---------------------------------------------------------------------------
// Kernel 3: persistent recurrent scan. grid=2 (blockIdx.x = direction),
// block = 512 thr (16 waves). Each wave owns one (batch-tile, u-group) unit
// with matched z/r/h tiles. All 12 recurrent-weight B fragments are hoisted
// into registers before the time loop; fp32 state lives in registers; LDS
// holds only a bf16 mirror of h for the cross-wave A fragments.
__global__ __launch_bounds__(512) void scan_kernel(
    const float* __restrict__ xg_f, const float* __restrict__ xg_b,
    const __bf16* __restrict__ Ut_f, const __bf16* __restrict__ Ut_b,
    const float* __restrict__ b1_f, const float* __restrict__ b1_b,
    float* __restrict__ hs) {
    __shared__ __bf16 lds_hb[B_][U_ + 8];      // bf16 h mirror, padded

    const int dir = blockIdx.x;
    const float*  __restrict__ xg = dir ? xg_b : xg_f;   // [T][B][384]
    const __bf16* __restrict__ Ut = dir ? Ut_b : Ut_f;   // [384][128] bf16(T)
    const float*  __restrict__ b1 = dir ? b1_b : b1_f;

    for (int i = threadIdx.x; i < B_ * (U_ + 8); i += 512)
        ((__bf16*)lds_hb)[i] = (__bf16)0.0f;
    __syncthreads();

    const int wave  = threadIdx.x >> 5;        // 0..15
    const int lane  = threadIdx.x & 31;
    const int rt    = wave & 1;                // batch rows rt*16..+16
    const int ug    = wave >> 1;               // u cols ug*16..+16
    const int nz = ug * 16, nr = U_ + ug * 16, nh = 2 * U_ + ug * 16;
    const int ncol  = lane & 15;
    const int mhalf = (lane >> 4) * 8;
    const int u     = ug * 16 + ncol;
    const float b1z = b1[nz + ncol], b1r = b1[nr + ncol], b1h = b1[nh + ncol];

    // loop-invariant recurrent weights: 12 B fragments kept in VGPRs
    v16bf wz[4], wr[4], wh[4];
#pragma unroll
    for (int kc = 0; kc < 4; ++kc) {
        wz[kc] = frag_b_tr(Ut + (size_t)nz * U_ + kc * 32, U_);
        wr[kc] = frag_b_tr(Ut + (size_t)nr * U_ + kc * 32, U_);
        wh[kc] = frag_b_tr(Ut + (size_t)nh * U_ + kc * 32, U_);
    }

    float hreg[8];                             // fp32 state, register-resident
#pragma unroll
    for (int r = 0; r < 8; ++r) hreg[r] = 0.0f;

    for (int s = 0; s < T_; ++s) {
        const int t = dir ? (T_ - 1 - s) : s;

        // prefetch next timestep's contiguous 49 KB xg block (one line/thread)
        if (s + 1 < T_ && threadIdx.x < (B_ * G3) / 32) {
            const int tn = dir ? (T_ - 2 - s) : (s + 1);
            __builtin_prefetch(
                (const void*)(xg + (size_t)tn * (B_ * G3) + threadIdx.x * 32), 0, 1);
        }

        // rec = h @ U for this wave's three matched tiles (z, r, h gates)
        v8f accz = zero8(), accr = zero8(), acch = zero8();
#pragma unroll
        for (int kc = 0; kc < 4; ++kc) {
            v16bf a = frag_a_bf16(&lds_hb[rt * 16][kc * 32], U_ + 8);
            accz = WMMA_BF16(a, wz[kc], accz);
            accr = WMMA_BF16(a, wr[kc], accr);
            acch = WMMA_BF16(a, wh[kc], acch);
        }
        __syncthreads();                       // all reads of lds_hb done

        // gates + state update, entirely per-lane
        const float* __restrict__ xt = xg + (size_t)t * (B_ * G3);
#pragma unroll
        for (int r = 0; r < 8; ++r) {
            const int b = rt * 16 + mhalf + r;
            const float xz = xt[b * G3 + nz + ncol];
            const float xr = xt[b * G3 + nr + ncol];
            const float xh = xt[b * G3 + nh + ncol];
            const float z  = sigmoidf_(xz + accz[r] + b1z);
            const float rg = sigmoidf_(xr + accr[r] + b1r);
            const float hh = fmaxf(0.0f, xh + rg * (acch[r] + b1h));
            const float hn = z * hreg[r] + (1.0f - z) * hh;
            hreg[r] = hn;
            lds_hb[b][u] = (__bf16)hn;
            hs[((size_t)t * B_ + b) * (2 * U_) + dir * U_ + u] = hn;
        }
        __syncthreads();                       // h update visible to all waves
    }
}

// ---------------------------------------------------------------------------
// Kernel 4: logits = h @ Wd + bd, softmax over C=11. One thread per (t,b) row
// of hs[T][B][2U]; output stays [b][t][c].
__global__ __launch_bounds__(256) void dense_softmax_kernel(
    const float* __restrict__ hs, const float* __restrict__ Wd,
    const float* __restrict__ bd, float* __restrict__ out) {
    __shared__ float lds_w[2 * U_ * C_];
    __shared__ float lds_b[C_];
    for (int i = threadIdx.x; i < 2 * U_ * C_; i += 256) lds_w[i] = Wd[i];
    if (threadIdx.x < C_) lds_b[threadIdx.x] = bd[threadIdx.x];
    __syncthreads();

    const size_t rr = (size_t)blockIdx.x * 256 + threadIdx.x;  // rr = t*B + b
    const int b = (int)(rr & (B_ - 1));
    const int t = (int)(rr >> 5);                              // B = 32 = 2^5
    const float4* __restrict__ hrow = (const float4*)(hs + rr * (2 * U_));

    float acc[C_];
#pragma unroll
    for (int c = 0; c < C_; ++c) acc[c] = lds_b[c];
    for (int k4 = 0; k4 < (2 * U_) / 4; ++k4) {
        const float4 hv = hrow[k4];
        const float* w0 = &lds_w[(4 * k4) * C_];
#pragma unroll
        for (int c = 0; c < C_; ++c)
            acc[c] += hv.x * w0[c] + hv.y * w0[C_ + c] +
                      hv.z * w0[2 * C_ + c] + hv.w * w0[3 * C_ + c];
    }
    float mx = acc[0];
#pragma unroll
    for (int c = 1; c < C_; ++c) mx = fmaxf(mx, acc[c]);
    float sum = 0.0f;
#pragma unroll
    for (int c = 0; c < C_; ++c) { acc[c] = __expf(acc[c] - mx); sum += acc[c]; }
    const float inv = 1.0f / sum;
#pragma unroll
    for (int c = 0; c < C_; ++c)
        out[((size_t)b * T_ + t) * C_ + c] = acc[c] * inv;
}

// ---------------------------------------------------------------------------
extern "C" void kernel_launch(void* const* d_in, const int* in_sizes, int n_in,
                              void* d_out, int out_size, void* d_ws, size_t ws_size,
                              hipStream_t stream) {
    const float* x   = (const float*)d_in[0];   // [32,2048,128]
    const float* W_f = (const float*)d_in[1];   // [128,384]
    const float* U_f = (const float*)d_in[2];   // [128,384]
    const float* b_f = (const float*)d_in[3];   // [2,384]
    const float* W_b = (const float*)d_in[4];
    const float* U_b = (const float*)d_in[5];
    const float* b_b = (const float*)d_in[6];
    const float* Wd  = (const float*)d_in[7];   // [256,11]
    const float* bd  = (const float*)d_in[8];   // [11]

    // workspace carve-up (~269 MB total)
    char* ws = (char*)d_ws;
    size_t off = 0;
    auto carve = [&](size_t bytes) -> void* {
        off = (off + 255) & ~(size_t)255;
        void* p = ws + off;
        off += bytes;
        return p;
    };
    __bf16* Wt_f = (__bf16*)carve((size_t)G3 * F_ * 2);
    __bf16* Wt_b = (__bf16*)carve((size_t)G3 * F_ * 2);
    __bf16* Ut_f = (__bf16*)carve((size_t)G3 * U_ * 2);
    __bf16* Ut_b = (__bf16*)carve((size_t)G3 * U_ * 2);
    float*  xg_f = (float*)carve((size_t)MROWS * G3 * 4);   // [T][B][384]
    float*  xg_b = (float*)carve((size_t)MROWS * G3 * 4);   // [T][B][384]
    float*  hsbf = (float*)carve((size_t)MROWS * 2 * U_ * 4); // [T][B][256]

    // 1) convert + transpose all four weight matrices to bf16 [N][K]
    const int wk = F_ * G3;
    cvt_tr_kernel<<<(wk + 255) / 256, 256, 0, stream>>>(W_f, Wt_f, F_, G3);
    cvt_tr_kernel<<<(wk + 255) / 256, 256, 0, stream>>>(W_b, Wt_b, F_, G3);
    cvt_tr_kernel<<<(wk + 255) / 256, 256, 0, stream>>>(U_f, Ut_f, U_, G3);
    cvt_tr_kernel<<<(wk + 255) / 256, 256, 0, stream>>>(U_b, Ut_b, U_, G3);

    // 2) input projections (WMMA GEMM), both directions, time-major output
    dim3 g2(MROWS / 64, G3 / 128, 1);
    proj_kernel<<<g2, 256, 0, stream>>>(x, Wt_f, b_f, xg_f);
    proj_kernel<<<g2, 256, 0, stream>>>(x, Wt_b, b_b, xg_b);

    // 3) both recurrent scans concurrently (blockIdx.x = direction)
    scan_kernel<<<2, 512, 0, stream>>>(xg_f, xg_b, Ut_f, Ut_b,
                                       b_f + G3, b_b + G3, hsbf);

    // 4) dense + softmax
    dense_softmax_kernel<<<MROWS / 256, 256, 0, stream>>>(hsbf, Wd, bd, (float*)d_out);
}
